// SparseBottleneck_84851373900101
// MI455X (gfx1250) — matrix operand order repaired and checked by reading, not compiled, not run on hardware
//
#include <hip/hip_runtime.h>
#include <hip/hip_bf16.h>

typedef __attribute__((ext_vector_type(16))) __bf16 v16bf;
typedef __attribute__((ext_vector_type(8)))  float  v8f;

#define B_SZ   16384
#define IN_DIM 4096
#define HID    2048
#define BOT    64
#define NCODES 8192

// ---------- helpers ----------
__device__ __forceinline__ unsigned short f32_to_bf16(float f) {
    unsigned int u = __float_as_uint(f);
    unsigned int r = u + 0x7FFFu + ((u >> 16) & 1u);   // round-to-nearest-even
    return (unsigned short)(r >> 16);
}

union Frag16 {           // 32 bytes: one WMMA 16-bit A/B operand (8 VGPRs)
    v16bf v;
    uint4 q[2];
};

// CDNA5 async global->LDS copy (ASYNCcnt-tracked), GV addressing mode.
__device__ __forceinline__ void async_b128(void* lds_ptr, const void* gptr) {
    unsigned loff = (unsigned)(size_t)lds_ptr;              // LDS aperture: offset = addr[31:0]
    unsigned long long ga = (unsigned long long)(size_t)gptr;
    asm volatile("global_load_async_to_lds_b128 %0, %1, off"
                 :: "v"(loff), "v"(ga) : "memory");
}
__device__ __forceinline__ void wait_async0() {
    asm volatile("s_wait_asynccnt 0x0" ::: "memory");
}

// ---------- elementwise f32 -> bf16 ----------
__global__ void cvt_bf16_kernel(const float* __restrict__ in,
                                unsigned short* __restrict__ out, int n) {
    int i = blockIdx.x * blockDim.x + threadIdx.x;
    if (i < n) out[i] = f32_to_bf16(in[i]);
}

// ---------- transpose + convert: in[R][C] f32 -> out[C][R] bf16 ----------
__global__ void transpose_cvt_kernel(const float* __restrict__ in,
                                     unsigned short* __restrict__ out,
                                     int R, int C) {
    __shared__ float tile[32][33];
    int bx = blockIdx.x * 32;   // col base in `in`
    int by = blockIdx.y * 32;   // row base in `in`
    int tx = threadIdx.x, ty = threadIdx.y;   // blockDim = (32, 8)
#pragma unroll
    for (int i = 0; i < 32; i += 8)
        tile[ty + i][tx] = in[(size_t)(by + ty + i) * C + bx + tx];
    __syncthreads();
#pragma unroll
    for (int i = 0; i < 32; i += 8)
        out[(size_t)(bx + ty + i) * R + by + tx] = f32_to_bf16(tile[tx][ty + i]);
}

// ---------- half squared norms of codebook rows ----------
__global__ void code_norm_kernel(const float* __restrict__ cb, float* __restrict__ nrm) {
    int c = blockIdx.x * blockDim.x + threadIdx.x;
    if (c < NCODES) {
        float s = 0.f;
#pragma unroll 8
        for (int k = 0; k < BOT; ++k) { float v = cb[(size_t)c * BOT + k]; s += v * v; }
        nrm[c] = 0.5f * s;
    }
}

// ---------- bf16 WMMA GEMM:  C[M][N] = act(A[M][K] @ Bt[N][K]^T + bias[N]) ----------
// Macro tile 256x64 (8 waves x 32 rows), BK=32, double-buffered LDS filled by
// async global->LDS copies. Each wave: 2 A-frags x 4 B-frags -> 8 WMMAs/K-step.
#define BM 256
#define BN 64
#define BK 32
__global__ __launch_bounds__(256) void gemm_bf16_kernel(
    const unsigned short* __restrict__ A, const unsigned short* __restrict__ Bt,
    const float* __restrict__ bias, float* __restrict__ Cf,
    unsigned short* __restrict__ Cbf, int M, int N, int K, int relu) {
    __shared__ __align__(16) unsigned short As[2][BM][BK];   // 2 x 16 KB
    __shared__ __align__(16) unsigned short Bs[2][BN][BK];   // 2 x  4 KB

    const int tid  = threadIdx.x;
    const int wave = tid >> 5;
    const int lane = tid & 31;
    const int half = lane >> 4;
    const int l16  = lane & 15;
    const int tileM = blockIdx.y * BM;
    const int tileN = blockIdx.x * BN;

    v8f acc0[4] = {};
    v8f acc1[4] = {};

    // stage one K-tile into LDS buffer `buf` via async copies (5 x b128 / thread)
    auto stage = [&](int buf, int kt) {
#pragma unroll
        for (int i = 0; i < 4; ++i) {                 // A: 1024 chunks / 256 thr
            int c = tid + i * 256, row = c >> 2, cc = c & 3;
            async_b128(&As[buf][row][cc * 8],
                       &A[(size_t)(tileM + row) * K + kt + cc * 8]);
        }
        int rb = tid >> 2, cb = tid & 3;              // B: 256 chunks
        async_b128(&Bs[buf][rb][cb * 8],
                   &Bt[(size_t)(tileN + rb) * K + kt + cb * 8]);
    };

    const int nk = K / BK;
    stage(0, 0);
    wait_async0();
    __syncthreads();

    for (int k = 0; k < nk; ++k) {
        const int buf = k & 1;
        if (k + 1 < nk) stage(buf ^ 1, (k + 1) * BK);   // prefetch next tile

        // A fragments: rows wave*32 + l16 and +16; K-runs at 8*half, 16+8*half
        Frag16 a0, a1;
        const int ar = wave * 32 + l16;
        a0.q[0] = *(const uint4*)&As[buf][ar][8 * half];
        a0.q[1] = *(const uint4*)&As[buf][ar][16 + 8 * half];
        a1.q[0] = *(const uint4*)&As[buf][ar + 16][8 * half];
        a1.q[1] = *(const uint4*)&As[buf][ar + 16][16 + 8 * half];

#pragma unroll
        for (int n = 0; n < 4; ++n) {
            Frag16 b;                                  // B frag: K-runs at 8*half, 16+8*half
            const int brow = n * 16 + l16;
            b.q[0] = *(const uint4*)&Bs[buf][brow][8 * half];
            b.q[1] = *(const uint4*)&Bs[buf][brow][16 + 8 * half];
            acc0[n] = __builtin_amdgcn_wmma_f32_16x16x32_bf16(
                false, a0.v, false, b.v, (short)0, acc0[n], false, false);
            acc1[n] = __builtin_amdgcn_wmma_f32_16x16x32_bf16(
                false, a1.v, false, b.v, (short)0, acc1[n], false, false);
        }
        wait_async0();          // next tile landed in LDS
        __syncthreads();        // all waves done reading current + writing next
    }

    // ---- epilogue: D layout lane L -> col L%16, rows v + 8*half ----
#pragma unroll
    for (int n = 0; n < 4; ++n) {
        const int col = tileN + n * 16 + l16;
        const float bv = bias ? bias[col] : 0.f;
#pragma unroll
        for (int i = 0; i < 2; ++i) {
            const v8f& acc = i ? acc1[n] : acc0[n];
#pragma unroll
            for (int v = 0; v < 8; ++v) {
                const int row = tileM + wave * 32 + i * 16 + v + 8 * half;
                float val = acc[v] + bv;
                if (relu) val = fmaxf(val, 0.f);
                if (Cf)  Cf [(size_t)row * N + col] = val;
                if (Cbf) Cbf[(size_t)row * N + col] = f32_to_bf16(val);
            }
        }
    }
}

// ---------- VQ: argmax_c (z.c - 0.5|c|^2) via WMMA, then gather z_q ----------
__global__ __launch_bounds__(256) void vq_kernel(
    const unsigned short* __restrict__ zeb,     // [B][64] bf16
    const unsigned short* __restrict__ cbb,     // [NCODES][64] bf16
    const float* __restrict__ cbf,              // [NCODES][64] f32
    const float* __restrict__ hnorm,            // [NCODES] 0.5|c|^2
    float* __restrict__ z_out,                  // [B][64] f32  (d_out)
    unsigned short* __restrict__ zq_bf) {       // [B][64] bf16
    const int tid  = threadIdx.x;
    const int wave = tid >> 5;
    const int lane = tid & 31;
    const int half = lane >> 4;
    const int l16  = lane & 15;
    const int base = (blockIdx.x * 8 + wave) * 16;   // 16 rows per wave

    // A fragments for this 16-row strip, K=64 -> two k-chunks
    Frag16 a0, a1;
    {
        const size_t r = (size_t)(base + l16) * BOT;
        a0.q[0] = *(const uint4*)&zeb[r + 8 * half];
        a0.q[1] = *(const uint4*)&zeb[r + 16 + 8 * half];
        a1.q[0] = *(const uint4*)&zeb[r + 32 + 8 * half];
        a1.q[1] = *(const uint4*)&zeb[r + 32 + 16 + 8 * half];
    }

    float best[8];
    int   bidx[8];
#pragma unroll
    for (int v = 0; v < 8; ++v) { best[v] = -INFINITY; bidx[v] = 0; }

    for (int nt = 0; nt < NCODES / 16; ++nt) {
        Frag16 b0, b1;
        const size_t r = (size_t)(nt * 16 + l16) * BOT;
        b0.q[0] = *(const uint4*)&cbb[r + 8 * half];
        b0.q[1] = *(const uint4*)&cbb[r + 16 + 8 * half];
        b1.q[0] = *(const uint4*)&cbb[r + 32 + 8 * half];
        b1.q[1] = *(const uint4*)&cbb[r + 32 + 16 + 8 * half];

        v8f acc = {};
        acc = __builtin_amdgcn_wmma_f32_16x16x32_bf16(false, a0.v, false, b0.v,
                                                      (short)0, acc, false, false);
        acc = __builtin_amdgcn_wmma_f32_16x16x32_bf16(false, a1.v, false, b1.v,
                                                      (short)0, acc, false, false);
        const int code = nt * 16 + l16;
        const float hn = hnorm[code];
#pragma unroll
        for (int v = 0; v < 8; ++v) {
            float s = acc[v] - hn;
            if (s > best[v]) { best[v] = s; bidx[v] = code; }
        }
    }

    // reduce across the 16 lanes that share `half` (columns of the D tile)
#pragma unroll
    for (int s = 1; s < 16; s <<= 1) {
#pragma unroll
        for (int v = 0; v < 8; ++v) {
            float ov = __shfl_xor(best[v], s, 32);
            int   oi = __shfl_xor(bidx[v], s, 32);
            if (ov > best[v] || (ov == best[v] && oi < bidx[v])) {
                best[v] = ov; bidx[v] = oi;
            }
        }
    }

    // gather winning codebook rows; all 32 lanes copy 64 floats per row
#pragma unroll
    for (int r = 0; r < 16; ++r) {
        const int code = __shfl(bidx[r & 7], (r >> 3) * 16, 32);
        const size_t orow = (size_t)(base + r) * BOT;
        const size_t crow = (size_t)code * BOT;
#pragma unroll
        for (int j = 0; j < 2; ++j) {
            const int k = lane + j * 32;
            const float val = cbf[crow + k];
            z_out[orow + k] = val;
            zq_bf[orow + k] = f32_to_bf16(val);
        }
    }
}

// ---------- host ----------
extern "C" void kernel_launch(void* const* d_in, const int* in_sizes, int n_in,
                              void* d_out, int out_size, void* d_ws, size_t ws_size,
                              hipStream_t stream) {
    const float* x   = (const float*)d_in[0];
    const float* W1  = (const float*)d_in[1];
    const float* b1  = (const float*)d_in[2];
    const float* W2  = (const float*)d_in[3];
    const float* b2  = (const float*)d_in[4];
    const float* cb  = (const float*)d_in[5];
    const float* dW1 = (const float*)d_in[6];
    const float* db1 = (const float*)d_in[7];
    const float* dW2 = (const float*)d_in[8];
    const float* db2 = (const float*)d_in[9];

    float* z_out = (float*)d_out;                          // [B][64]
    float* xrec  = (float*)d_out + (size_t)B_SZ * BOT;     // [B][IN_DIM]

    char* w = (char*)d_ws;
    auto alloc = [&](size_t bytes) {
        char* p = w; w += (bytes + 255) & ~(size_t)255; return p;
    };
    unsigned short* x_bf   = (unsigned short*)alloc((size_t)B_SZ * IN_DIM * 2);
    unsigned short* W1t    = (unsigned short*)alloc((size_t)HID * IN_DIM * 2);
    unsigned short* W2t    = (unsigned short*)alloc((size_t)BOT * HID * 2);
    unsigned short* dW1t   = (unsigned short*)alloc((size_t)HID * BOT * 2);
    unsigned short* dW2t   = (unsigned short*)alloc((size_t)IN_DIM * HID * 2);
    unsigned short* cb_bf  = (unsigned short*)alloc((size_t)NCODES * BOT * 2);
    float*          cnorm  = (float*)alloc((size_t)NCODES * 4);
    unsigned short* h_bf   = (unsigned short*)alloc((size_t)B_SZ * HID * 2);
    unsigned short* ze_bf  = (unsigned short*)alloc((size_t)B_SZ * BOT * 2);
    unsigned short* zq_bf  = (unsigned short*)alloc((size_t)B_SZ * BOT * 2);
    unsigned short* h2_bf  = (unsigned short*)alloc((size_t)B_SZ * HID * 2);

    // 1) precision conversion / weight transposes
    {
        int n = B_SZ * IN_DIM;
        cvt_bf16_kernel<<<(n + 255) / 256, 256, 0, stream>>>(x, x_bf, n);
        n = NCODES * BOT;
        cvt_bf16_kernel<<<(n + 255) / 256, 256, 0, stream>>>(cb, cb_bf, n);
        dim3 tb(32, 8);
        transpose_cvt_kernel<<<dim3(HID / 32, IN_DIM / 32), tb, 0, stream>>>(W1,  W1t,  IN_DIM, HID);
        transpose_cvt_kernel<<<dim3(BOT / 32, HID / 32),    tb, 0, stream>>>(W2,  W2t,  HID,    BOT);
        transpose_cvt_kernel<<<dim3(HID / 32, BOT / 32),    tb, 0, stream>>>(dW1, dW1t, BOT,    HID);
        transpose_cvt_kernel<<<dim3(IN_DIM / 32, HID / 32), tb, 0, stream>>>(dW2, dW2t, HID,    IN_DIM);
        code_norm_kernel<<<(NCODES + 255) / 256, 256, 0, stream>>>(cb, cnorm);
    }

    // 2) h = relu(x @ W1 + b1)                     [B][HID] bf16
    gemm_bf16_kernel<<<dim3(HID / BN, B_SZ / BM), 256, 0, stream>>>(
        x_bf, W1t, b1, nullptr, h_bf, B_SZ, HID, IN_DIM, 1);

    // 3) z_e = h @ W2 + b2                         [B][BOT] bf16
    gemm_bf16_kernel<<<dim3(BOT / BN, B_SZ / BM), 256, 0, stream>>>(
        h_bf, W2t, b2, nullptr, ze_bf, B_SZ, BOT, HID, 0);

    // 4) VQ argmin + gather: z (= z_q) f32 to d_out, bf16 copy for decoder
    vq_kernel<<<B_SZ / (16 * 8), 256, 0, stream>>>(
        ze_bf, cb_bf, cb, cnorm, z_out, zq_bf);

    // 5) h2 = relu(z_q @ dW1 + db1)                [B][HID] bf16
    gemm_bf16_kernel<<<dim3(HID / BN, B_SZ / BM), 256, 0, stream>>>(
        zq_bf, dW1t, db1, nullptr, h2_bf, B_SZ, HID, BOT, 1);

    // 6) x_recon = h2 @ dW2 + db2                  [B][IN_DIM] f32 to d_out
    gemm_bf16_kernel<<<dim3(IN_DIM / BN, B_SZ / BM), 256, 0, stream>>>(
        h2_bf, dW2t, db2, xrec, nullptr, B_SZ, IN_DIM, HID, 0);
}
